// SimplifiedGSAAttention_31817117729243
// MI455X (gfx1250) — matrix operand order repaired and checked by reading, not compiled, not run on hardware
//
#include <hip/hip_runtime.h>

// ---------------------------------------------------------------------------
// SimplifiedGSAAttention for MI455X (gfx1250, wave32, WMMA).
// Compute-bound (workspace < L2 192MB): v_wmma_f32_16x16x32_bf16, f32 accum.
// Gaussian-center scores = affine transform of ONE shared QK^T per head; the
// 4 per-center softmaxes are imp-combined BEFORE a single PV matmul.
// This round: ping-pong double buffering (two fragment register sets, no
// rotation moves) in the GEMM and attention score loops.
// ---------------------------------------------------------------------------

#define DEV __device__ __forceinline__

typedef __attribute__((ext_vector_type(16))) __bf16 v16bf;
typedef __attribute__((ext_vector_type(8)))  float  v8f;

union FragU { unsigned int u[8]; v16bf v; };

DEV unsigned short f2bf(float f) {
  union { float f; unsigned int u; } c; c.f = f;
  unsigned int r = c.u + 0x7FFFu + ((c.u >> 16) & 1u);   // round-to-nearest-even
  return (unsigned short)(r >> 16);
}
DEV unsigned int pack2bf(float a, float b) {
  return (unsigned int)f2bf(a) | ((unsigned int)f2bf(b) << 16);
}
DEV v8f zero8() {
  v8f z;
#pragma unroll
  for (int i = 0; i < 8; ++i) z[i] = 0.0f;
  return z;
}

// A fragment, 16x32 (MxK) bf16, row-major source: element (m,k) at base[m*ld+k].
// contiguous k order (k = 16*half + s), matching load_frag_b; per-lane = one contiguous 32B run.
DEV v16bf load_frag_a(const unsigned short* base, int ld, int lane) {
  const int m = lane & 15, half = lane >> 4;
  const unsigned short* row = base + m * ld;
  FragU f;
#pragma unroll
  for (int v = 0; v < 8; ++v) {
    int k = 2 * v + 16 * half;
    f.u[v] = *(const unsigned int*)(row + k);
  }
  return f.v;
}

// B fragment, 32x16 (KxN) bf16, N-major source: element (k,n) at baseT[n*ld+k].
DEV v16bf load_frag_b(const unsigned short* baseT, int ld, int lane) {
  const int n = lane & 15, half = lane >> 4;
  const unsigned short* col = baseT + n * ld + 16 * half;
  FragU f;
#pragma unroll
  for (int v = 0; v < 8; ++v) f.u[v] = *(const unsigned int*)(col + 2 * v);
  return f.v;
}

DEV v8f wmma_bf16(v16bf a, v16bf b, v8f c) {
  return __builtin_amdgcn_wmma_f32_16x16x32_bf16(false, a, false, b, (short)0, c, false, false);
}

// ---------------------------------------------------------------------------
// prep kernels
// ---------------------------------------------------------------------------
__global__ void k_cvt_bf16(const float* __restrict__ src,
                           unsigned short* __restrict__ dst, int n) {
  int i = blockIdx.x * blockDim.x + threadIdx.x;
  if (i < n) dst[i] = f2bf(src[i]);
}

__global__ void k_transpose_bf16(const float* __restrict__ W,
                                 unsigned short* __restrict__ Wt,
                                 int rows, int cols) {
  int i = blockIdx.x * blockDim.x + threadIdx.x;
  if (i >= rows * cols) return;
  int r = i / cols, c = i % cols;
  Wt[(size_t)c * rows + r] = f2bf(W[i]);
}

// ---------------------------------------------------------------------------
// bf16 WMMA GEMM + bias:  C[MxN] = A[MxK] * Bt[NxK]^T + bias.
// One wave owns a 32x64 tile.  Ping-pong double buffer over 64-wide k-chunks:
// set Y loads are issued before set X's 8 WMMAs and vice versa -> no rotation
// moves, loads overlap the XDL pipe.  K must be a multiple of 64.
// ---------------------------------------------------------------------------
__global__ __launch_bounds__(128, 1) void k_gemm_bf16_bias(
    const unsigned short* __restrict__ A,
    const unsigned short* __restrict__ Bt,
    const float* __restrict__ bias,
    float* __restrict__ C,
    int M, int N, int K) {
  const int lane = threadIdx.x & 31;
  const int wid = (blockIdx.x * blockDim.x + threadIdx.x) >> 5;
  const int ntn = N >> 6;
  if (wid >= (M >> 5) * ntn) return;
  const int mt = wid / ntn, nt = wid % ntn;
  const int m0 = mt * 32, n0 = nt * 64;

  const unsigned short* Arow0 = A + (size_t)(m0 + 0)  * K;
  const unsigned short* Arow1 = A + (size_t)(m0 + 16) * K;
  const unsigned short* Bcol0 = Bt + (size_t)(n0 +  0) * K;
  const unsigned short* Bcol1 = Bt + (size_t)(n0 + 16) * K;
  const unsigned short* Bcol2 = Bt + (size_t)(n0 + 32) * K;
  const unsigned short* Bcol3 = Bt + (size_t)(n0 + 48) * K;

  v8f acc[2][4];
#pragma unroll
  for (int i = 0; i < 2; ++i)
#pragma unroll
    for (int t = 0; t < 4; ++t) acc[i][t] = zero8();

  // set X: chunk k0 ; set Y: chunk k0+32
  v16bf xa0 = load_frag_a(Arow0, K, lane);
  v16bf xa1 = load_frag_a(Arow1, K, lane);
  v16bf xb0 = load_frag_b(Bcol0, K, lane);
  v16bf xb1 = load_frag_b(Bcol1, K, lane);
  v16bf xb2 = load_frag_b(Bcol2, K, lane);
  v16bf xb3 = load_frag_b(Bcol3, K, lane);

  for (int k0 = 0; k0 < K; k0 += 64) {
    const int ky = k0 + 32;                 // always < K (K % 64 == 0)
    __builtin_prefetch(Arow0 + k0 + 128, 0, 1);      // global_prefetch_b8
    v16bf ya0 = load_frag_a(Arow0 + ky, K, lane);
    v16bf ya1 = load_frag_a(Arow1 + ky, K, lane);
    v16bf yb0 = load_frag_b(Bcol0 + ky, K, lane);
    v16bf yb1 = load_frag_b(Bcol1 + ky, K, lane);
    v16bf yb2 = load_frag_b(Bcol2 + ky, K, lane);
    v16bf yb3 = load_frag_b(Bcol3 + ky, K, lane);

    acc[0][0] = wmma_bf16(xa0, xb0, acc[0][0]);
    acc[1][0] = wmma_bf16(xa1, xb0, acc[1][0]);
    acc[0][1] = wmma_bf16(xa0, xb1, acc[0][1]);
    acc[1][1] = wmma_bf16(xa1, xb1, acc[1][1]);
    acc[0][2] = wmma_bf16(xa0, xb2, acc[0][2]);
    acc[1][2] = wmma_bf16(xa1, xb2, acc[1][2]);
    acc[0][3] = wmma_bf16(xa0, xb3, acc[0][3]);
    acc[1][3] = wmma_bf16(xa1, xb3, acc[1][3]);

    const int kx = k0 + 64;
    if (kx < K) {                           // reload set X for next iteration
      xa0 = load_frag_a(Arow0 + kx, K, lane);
      xa1 = load_frag_a(Arow1 + kx, K, lane);
      xb0 = load_frag_b(Bcol0 + kx, K, lane);
      xb1 = load_frag_b(Bcol1 + kx, K, lane);
      xb2 = load_frag_b(Bcol2 + kx, K, lane);
      xb3 = load_frag_b(Bcol3 + kx, K, lane);
    }
    acc[0][0] = wmma_bf16(ya0, yb0, acc[0][0]);
    acc[1][0] = wmma_bf16(ya1, yb0, acc[1][0]);
    acc[0][1] = wmma_bf16(ya0, yb1, acc[0][1]);
    acc[1][1] = wmma_bf16(ya1, yb1, acc[1][1]);
    acc[0][2] = wmma_bf16(ya0, yb2, acc[0][2]);
    acc[1][2] = wmma_bf16(ya1, yb2, acc[1][2]);
    acc[0][3] = wmma_bf16(ya0, yb3, acc[0][3]);
    acc[1][3] = wmma_bf16(ya1, yb3, acc[1][3]);
  }

  const int n16 = lane & 15, half = lane >> 4;
#pragma unroll
  for (int i = 0; i < 2; ++i)
#pragma unroll
    for (int t = 0; t < 4; ++t)
#pragma unroll
      for (int j = 0; j < 8; ++j) {
        int row = m0 + i * 16 + j + 8 * half;
        int col = n0 + t * 16 + n16;
        C[(size_t)row * N + col] = acc[i][t][j] + bias[col];
      }
}

// ---------------------------------------------------------------------------
// LayerNorm of q,k over Dh=64 (one wave per (b,h,t) row) + bf16 Q/K in
// [bh][t][d], V transposed to [bh][d][t], center-dot corrections q.c_s / k.c_s.
// ---------------------------------------------------------------------------
__global__ __launch_bounds__(128, 1) void k_ln_prep(
    const float* __restrict__ qkv,
    const float* __restrict__ ln_g, const float* __restrict__ ln_b,
    const float* __restrict__ centers,
    unsigned short* __restrict__ Qb, unsigned short* __restrict__ Kb,
    unsigned short* __restrict__ Vt,
    float* __restrict__ qdot, float* __restrict__ kdot) {
  const int lane = threadIdx.x & 31;
  const int row = (blockIdx.x * blockDim.x + threadIdx.x) >> 5;  // [0, B*H*T)
  const int bh = row >> 10, t = row & 1023;
  const int b = bh >> 4, h = bh & 15;
  const float* base = qkv + ((size_t)(b * 1024 + t)) * 3072 + h * 64;
  const int d0 = lane * 2, d1 = d0 + 1;
  const float g0 = ln_g[d0], g1 = ln_g[d1], lb0 = ln_b[d0], lb1 = ln_b[d1];

#pragma unroll
  for (int which = 0; which < 2; ++which) {
    const float* src = base + which * 1024;
    float x0 = src[d0], x1 = src[d1];
    float sum = x0 + x1;
#pragma unroll
    for (int m = 1; m < 32; m <<= 1) sum += __shfl_xor(sum, m, 32);
    float mu = sum * (1.0f / 64.0f);
    float e0 = x0 - mu, e1 = x1 - mu;
    float vv = e0 * e0 + e1 * e1;
#pragma unroll
    for (int m = 1; m < 32; m <<= 1) vv += __shfl_xor(vv, m, 32);
    float rstd = rsqrtf(vv * (1.0f / 64.0f) + 1e-5f);
    float y0 = e0 * rstd * g0 + lb0;
    float y1 = e1 * rstd * g1 + lb1;
    unsigned short* dst = (which == 0 ? Qb : Kb) + ((size_t)(bh * 1024 + t)) * 64 + d0;
    *(unsigned int*)dst = pack2bf(y0, y1);
    float* ddst = (which == 0 ? qdot : kdot);
#pragma unroll
    for (int s = 0; s < 4; ++s) {
      const float* c = centers + (h * 4 + s) * 64;
      float pz = y0 * c[d0] + y1 * c[d1];
#pragma unroll
      for (int m = 1; m < 32; m <<= 1) pz += __shfl_xor(pz, m, 32);
      if (lane == 0) ddst[(size_t)(bh * 4 + s) * 1024 + t] = pz;
    }
  }
  const float* vsrc = base + 2048;
  Vt[(size_t)(bh * 64 + d0) * 1024 + t] = f2bf(vsrc[d0]);
  Vt[(size_t)(bh * 64 + d1) * 1024 + t] = f2bf(vsrc[d1]);
}

// ---------------------------------------------------------------------------
// Flash-style GSA attention, one wave per 16-row tile of one (b,h).
// scores_s = a_s*qk + rowc_s(t) + colc_s(u) + 0.01*amp_s*delta(t,u).
// Pass1: row max; Pass2: sumexp (both ping-pong double-buffered over u);
// Pass3: combined P (bf16 via LDS relayout) + single P@V accumulation with
// V-fragments hoisted above the exp/VALU block.
// ---------------------------------------------------------------------------
__global__ __launch_bounds__(128, 1) void k_attn(
    const unsigned short* __restrict__ Qb, const unsigned short* __restrict__ Kb,
    const unsigned short* __restrict__ Vt,
    const float* __restrict__ qdot, const float* __restrict__ kdot,
    const float* __restrict__ centers, const float* __restrict__ log_scales,
    const float* __restrict__ amplitudes,
    unsigned short* __restrict__ comb) {
  const int lane = threadIdx.x & 31;
  const int widx = threadIdx.x >> 5;
  const int w = blockIdx.x * 4 + widx;
  const int bh = w >> 6, tt = w & 63, t0 = tt * 16;
  const int h = bh & 15, b = bh >> 4;
  const int n16 = lane & 15, half = lane >> 4;

  // per-(h,s) scalars: a_s = amp/scale^2, diag term, imp = softmax_s(amp)
  float a_s[4], d01[4], impv[4];
  {
    float amp[4], mx = -3.0e38f;
#pragma unroll
    for (int s = 0; s < 4; ++s) { amp[s] = amplitudes[h * 4 + s]; mx = fmaxf(mx, amp[s]); }
    float se = 0.f;
#pragma unroll
    for (int s = 0; s < 4; ++s) { impv[s] = __expf(amp[s] - mx); se += impv[s]; }
    float rse = 1.0f / se;
#pragma unroll
    for (int s = 0; s < 4; ++s) {
      impv[s] *= rse;
      float sc = __expf(log_scales[h * 4 + s]);
      sc = fminf(fmaxf(sc, 1e-4f), 10.0f);
      a_s[s] = amp[s] / (sc * sc);
      d01[s] = 0.01f * amp[s];
    }
  }
  // row constants: a_s*(c.c - q_t.c)
  float rowc[8][4];
#pragma unroll
  for (int s = 0; s < 4; ++s) {
    const float* c = centers + (h * 4 + s) * 64;
    float cc = 0.f;
    for (int d = 0; d < 64; ++d) cc += c[d] * c[d];
#pragma unroll
    for (int j = 0; j < 8; ++j) {
      int tg = t0 + j + 8 * half;
      rowc[j][s] = a_s[s] * (cc - qdot[(size_t)(bh * 4 + s) * 1024 + tg]);
    }
  }

  const unsigned short* Qrow  = Qb + (size_t)(bh * 1024 + t0) * 64;
  const unsigned short* Kbase = Kb + (size_t)bh * 1024 * 64;
  const unsigned short* Vbase = Vt + (size_t)bh * 64 * 1024;
  const float* kdb = kdot + (size_t)bh * 4 * 1024;
  v16bf aq0 = load_frag_a(Qrow,      64, lane);
  v16bf aq1 = load_frag_a(Qrow + 32, 64, lane);

  float pmax[8][4], psum[8][4];
#pragma unroll
  for (int j = 0; j < 8; ++j)
#pragma unroll
    for (int s = 0; s < 4; ++s) { pmax[j][s] = -3.0e38f; psum[j][s] = 0.f; }

  // z for element (j, u=ug) of center s
  auto zval = [&](float qk, int j, int s, int ug, float colv) -> float {
    float dg = ((t0 + j + 8 * half) == ug) ? 1.0f : 0.0f;
    return fmaf(a_s[s], qk, rowc[j][s]) + colv + dg * d01[s];
  };

  // ---- pass 1: row maxima (ping-pong over 32-u chunks) ----
  {
    v16bf xb0 = load_frag_b(Kbase,      64, lane);
    v16bf xb1 = load_frag_b(Kbase + 32, 64, lane);
    for (int u0 = 0; u0 < 1024; u0 += 32) {
      const int uy = u0 + 16;               // always < 1024
      v16bf yb0 = load_frag_b(Kbase + (size_t)uy * 64,      64, lane);
      v16bf yb1 = load_frag_b(Kbase + (size_t)uy * 64 + 32, 64, lane);
      v8f c = zero8();
      c = wmma_bf16(aq0, xb0, c);
      c = wmma_bf16(aq1, xb1, c);
      {
        const int ug = u0 + n16;
        float colv[4];
#pragma unroll
        for (int s = 0; s < 4; ++s) colv[s] = -a_s[s] * kdb[s * 1024 + ug];
#pragma unroll
        for (int j = 0; j < 8; ++j)
#pragma unroll
          for (int s = 0; s < 4; ++s)
            pmax[j][s] = fmaxf(pmax[j][s], zval(c[j], j, s, ug, colv[s]));
      }
      const int ux = u0 + 32;
      if (ux < 1024) {
        xb0 = load_frag_b(Kbase + (size_t)ux * 64,      64, lane);
        xb1 = load_frag_b(Kbase + (size_t)ux * 64 + 32, 64, lane);
      }
      v8f cy = zero8();
      cy = wmma_bf16(aq0, yb0, cy);
      cy = wmma_bf16(aq1, yb1, cy);
      {
        const int ug = uy + n16;
        float colv[4];
#pragma unroll
        for (int s = 0; s < 4; ++s) colv[s] = -a_s[s] * kdb[s * 1024 + ug];
#pragma unroll
        for (int j = 0; j < 8; ++j)
#pragma unroll
          for (int s = 0; s < 4; ++s)
            pmax[j][s] = fmaxf(pmax[j][s], zval(cy[j], j, s, ug, colv[s]));
      }
    }
  }
#pragma unroll
  for (int j = 0; j < 8; ++j)
#pragma unroll
    for (int s = 0; s < 4; ++s)
#pragma unroll
      for (int m = 1; m < 16; m <<= 1)
        pmax[j][s] = fmaxf(pmax[j][s], __shfl_xor(pmax[j][s], m, 32));

  // ---- pass 2: sum of exp (ping-pong) ----
  {
    v16bf xb0 = load_frag_b(Kbase,      64, lane);
    v16bf xb1 = load_frag_b(Kbase + 32, 64, lane);
    for (int u0 = 0; u0 < 1024; u0 += 32) {
      const int uy = u0 + 16;
      v16bf yb0 = load_frag_b(Kbase + (size_t)uy * 64,      64, lane);
      v16bf yb1 = load_frag_b(Kbase + (size_t)uy * 64 + 32, 64, lane);
      v8f c = zero8();
      c = wmma_bf16(aq0, xb0, c);
      c = wmma_bf16(aq1, xb1, c);
      {
        const int ug = u0 + n16;
        float colv[4];
#pragma unroll
        for (int s = 0; s < 4; ++s) colv[s] = -a_s[s] * kdb[s * 1024 + ug];
#pragma unroll
        for (int j = 0; j < 8; ++j)
#pragma unroll
          for (int s = 0; s < 4; ++s)
            psum[j][s] += __expf(zval(c[j], j, s, ug, colv[s]) - pmax[j][s]);
      }
      const int ux = u0 + 32;
      if (ux < 1024) {
        xb0 = load_frag_b(Kbase + (size_t)ux * 64,      64, lane);
        xb1 = load_frag_b(Kbase + (size_t)ux * 64 + 32, 64, lane);
      }
      v8f cy = zero8();
      cy = wmma_bf16(aq0, yb0, cy);
      cy = wmma_bf16(aq1, yb1, cy);
      {
        const int ug = uy + n16;
        float colv[4];
#pragma unroll
        for (int s = 0; s < 4; ++s) colv[s] = -a_s[s] * kdb[s * 1024 + ug];
#pragma unroll
        for (int j = 0; j < 8; ++j)
#pragma unroll
          for (int s = 0; s < 4; ++s)
            psum[j][s] += __expf(zval(cy[j], j, s, ug, colv[s]) - pmax[j][s]);
      }
    }
  }
  float wjs[8][4];
#pragma unroll
  for (int j = 0; j < 8; ++j)
#pragma unroll
    for (int s = 0; s < 4; ++s) {
#pragma unroll
      for (int m = 1; m < 16; m <<= 1) psum[j][s] += __shfl_xor(psum[j][s], m, 32);
      wjs[j][s] = impv[s] / psum[j][s];
    }

  // ---- pass 3: combined P (bf16 via LDS relayout) then P @ V ----
  __shared__ unsigned short Plds[4][16][32];
  unsigned short(*P)[32] = Plds[widx];
  v8f o[4];
#pragma unroll
  for (int dt = 0; dt < 4; ++dt) o[dt] = zero8();

  for (int u0 = 0; u0 < 1024; u0 += 32) {
    // V-tile fragments first: overlap their latency with the exp/VALU block
    v16bf bv0 = load_frag_b(Vbase + (size_t)( 0) * 1024 + u0, 1024, lane);
    v16bf bv1 = load_frag_b(Vbase + (size_t)(16) * 1024 + u0, 1024, lane);
    v16bf bv2 = load_frag_b(Vbase + (size_t)(32) * 1024 + u0, 1024, lane);
    v16bf bv3 = load_frag_b(Vbase + (size_t)(48) * 1024 + u0, 1024, lane);
#pragma unroll
    for (int uh = 0; uh < 2; ++uh) {
      const int ub = u0 + uh * 16;
      v16bf b0 = load_frag_b(Kbase + (size_t)ub * 64,      64, lane);
      v16bf b1 = load_frag_b(Kbase + (size_t)ub * 64 + 32, 64, lane);
      v8f c = zero8();
      c = wmma_bf16(aq0, b0, c);
      c = wmma_bf16(aq1, b1, c);
      const int ug = ub + n16;
      float colv[4];
#pragma unroll
      for (int s = 0; s < 4; ++s) colv[s] = -a_s[s] * kdb[s * 1024 + ug];
#pragma unroll
      for (int j = 0; j < 8; ++j) {
        float pv = 0.f;
#pragma unroll
        for (int s = 0; s < 4; ++s)
          pv += wjs[j][s] * __expf(zval(c[j], j, s, ug, colv[s]) - pmax[j][s]);
        P[j + 8 * half][uh * 16 + n16] = f2bf(pv);   // C-layout -> row-major LDS
      }
    }
    // re-read as A fragment (per-wave LDS; DS ops are in-order within a wave)
    v16bf ap = load_frag_a(&P[0][0], 32, lane);
    o[0] = wmma_bf16(ap, bv0, o[0]);
    o[1] = wmma_bf16(ap, bv1, o[1]);
    o[2] = wmma_bf16(ap, bv2, o[2]);
    o[3] = wmma_bf16(ap, bv3, o[3]);
  }
  // write imp-combined head output, already in [b][t][h*64+d] layout (bf16)
#pragma unroll
  for (int dt = 0; dt < 4; ++dt)
#pragma unroll
    for (int j = 0; j < 8; ++j) {
      int tg = t0 + j + 8 * half;
      int col = h * 64 + dt * 16 + n16;
      comb[(size_t)(b * 1024 + tg) * 1024 + col] = f2bf(o[dt][j]);
    }
}

// ---------------------------------------------------------------------------
extern "C" void kernel_launch(void* const* d_in, const int* in_sizes, int n_in,
                              void* d_out, int out_size, void* d_ws, size_t ws_size,
                              hipStream_t stream) {
  (void)in_sizes; (void)n_in; (void)out_size; (void)ws_size;
  const float* x          = (const float*)d_in[0];
  const float* Wqkv       = (const float*)d_in[1];
  const float* bqkv       = (const float*)d_in[2];
  const float* Wout       = (const float*)d_in[3];
  const float* bout       = (const float*)d_in[4];
  const float* ln_g       = (const float*)d_in[5];
  const float* ln_b       = (const float*)d_in[6];
  const float* centers    = (const float*)d_in[7];
  const float* log_scales = (const float*)d_in[8];
  const float* amplitudes = (const float*)d_in[9];
  float* out = (float*)d_out;

  // workspace carve-out (~53 MiB total)
  char* p = (char*)d_ws;
  auto alloc = [&](size_t bytes) -> char* {
    char* r = p;
    p += (bytes + 255) & ~(size_t)255;
    return r;
  };
  unsigned short* Xb    = (unsigned short*)alloc(2048ull * 1024 * 2);
  unsigned short* WqkvT = (unsigned short*)alloc(3072ull * 1024 * 2);
  unsigned short* WoutT = (unsigned short*)alloc(1024ull * 1024 * 2);
  float*          qkv   = (float*)alloc(2048ull * 3072 * 4);
  unsigned short* Qb    = (unsigned short*)alloc(2048ull * 1024 * 2);
  unsigned short* Kb    = (unsigned short*)alloc(2048ull * 1024 * 2);
  unsigned short* Vt    = (unsigned short*)alloc(2048ull * 1024 * 2);
  float*          qdot  = (float*)alloc(2ull * 16 * 4 * 1024 * 4);
  float*          kdot  = (float*)alloc(2ull * 16 * 4 * 1024 * 4);
  unsigned short* comb  = (unsigned short*)alloc(2048ull * 1024 * 2);

  // 1) bf16 conversions / weight transposes
  k_cvt_bf16<<<(2048 * 1024) / 256, 256, 0, stream>>>(x, Xb, 2048 * 1024);
  k_transpose_bf16<<<(1024 * 3072) / 256, 256, 0, stream>>>(Wqkv, WqkvT, 1024, 3072);
  k_transpose_bf16<<<(1024 * 1024) / 256, 256, 0, stream>>>(Wout, WoutT, 1024, 1024);

  // 2) QKV GEMM: (2048 x 1024) * (1024 x 3072) + bqkv -> 3072 waves / 768 blocks
  k_gemm_bf16_bias<<<768, 128, 0, stream>>>(Xb, WqkvT, bqkv, qkv, 2048, 3072, 1024);

  // 3) LayerNorm + center dots + V transpose: one wave per (b,h,t) row
  k_ln_prep<<<(2 * 16 * 1024) / 4, 128, 0, stream>>>(qkv, ln_g, ln_b, centers,
                                                     Qb, Kb, Vt, qdot, kdot);

  // 4) attention: one wave per 16-row tile -> 2048 waves / 512 blocks
  k_attn<<<2048 / 4, 128, 0, stream>>>(Qb, Kb, Vt, qdot, kdot,
                                       centers, log_scales, amplitudes, comb);

  // 5) output GEMM: (2048 x 1024) * (1024 x 1024) + bout -> 1024 waves / 256 blocks
  k_gemm_bf16_bias<<<256, 128, 0, stream>>>(comb, WoutT, bout, out, 2048, 1024, 1024);
}
